// WindowedSelfAttention_64183991271548
// MI455X (gfx1250) — matrix operand order repaired and checked by reading, not compiled, not run on hardware
//
#include <hip/hip_runtime.h>
#include <hip/hip_bf16.h>
#include <cstddef>

// ---------------------------------------------------------------------------
// Types for CDNA5 WMMA (wave32, 16x16x32 f16 -> f32)
// ---------------------------------------------------------------------------
typedef __attribute__((ext_vector_type(16))) _Float16 v16h;
typedef __attribute__((ext_vector_type(8)))  _Float16 v8h;
typedef __attribute__((ext_vector_type(4)))  _Float16 v4h;
typedef __attribute__((ext_vector_type(8)))  float    v8f;

union Frag16 { v16h v; v8h h[2]; };

__device__ __forceinline__ v8f wmma_f16(v16h a, v16h b, v8f c) {
  // D = A(16x32) * B(32x16) + C(16x16 f32)
  return __builtin_amdgcn_wmma_f32_16x16x32_f16(
      /*neg_a=*/false, a, /*neg_b=*/false, b,
      /*c_mod=*/(short)0, c, /*reuse_a=*/false, /*reuse_b=*/false);
}

// A-fragment (16x32, row-major f16 source, leading dim ld).
// Lane: m = m0 + (lane&15), half = lane>>4.
// a[0..7]  = K in [k0 + half*8,      +8)
// a[8..15] = K in [k0 + 16 + half*8, +8)
__device__ __forceinline__ v16h load_a_f16(const _Float16* base, int ld,
                                           int m0, int k0, int lane) {
  int m    = m0 + (lane & 15);
  int half = lane >> 4;
  const _Float16* p = base + (size_t)m * ld + k0 + half * 8;
  Frag16 f;
  f.h[0] = *(const v8h*)p;
  f.h[1] = *(const v8h*)(p + 16);
  return f.v;
}

// B-fragment (32x16, row-major f16 source [K][ld]).
// Lane: n = n0 + (lane&15), half = lane>>4; b[0..7] = K [k0+half*8,+8), b[8..15] = K [k0+16+half*8,+8).
__device__ __forceinline__ v16h load_b_f16(const _Float16* base, int ld,
                                           int k0, int n0, int lane) {
  int half = lane >> 4;
  const _Float16* p = base + (size_t)(k0 + half * 8) * ld + n0 + (lane & 15);
  v16h f;
#pragma unroll
  for (int i = 0; i < 8; ++i) {
    f[i]     = p[(size_t)i * ld];
    f[8 + i] = p[(size_t)(16 + i) * ld];
  }
  return f;
}

// ---------------------------------------------------------------------------
// Problem constants (fixed by the reference harness)
// ---------------------------------------------------------------------------
// B=2, T=16, H=32, W=32, C=512, heads=8, head_dim=64, window 8x8x8
// windows: 64 total (b, nt(2), nh(4), nw(4)); 512 tokens/window (t,h,w)
__device__ __forceinline__ int global_token(int win, int l) {
  int b  = win >> 5, r = win & 31;
  int nt = r >> 4, nh = (r >> 2) & 3, nw = r & 3;
  int t  = l >> 6, h  = (l >> 3) & 7, w  = l & 7;
  return b * 16384 + (nt * 8 + t) * 1024 + (nh * 8 + h) * 32 + (nw * 8 + w);
}

// Workspace layout (bytes)
static constexpr size_t WQ_OFF = 0;                                      // 512*1536 f16
static constexpr size_t WP_OFF = WQ_OFF + (size_t)512 * 1536 * 2;        // 512*512 f16
static constexpr size_t XW_OFF = WP_OFF + (size_t)512 * 512 * 2;         // [64][512][512] f16
static constexpr size_t Q_OFF  = XW_OFF + (size_t)64 * 512 * 512 * 2;    // [64*8][512][64]
static constexpr size_t KT_OFF = Q_OFF  + (size_t)64 * 8 * 512 * 64 * 2; // [64*8][64][512]
static constexpr size_t V_OFF  = KT_OFF + (size_t)64 * 8 * 512 * 64 * 2; // [64*8][512][64]
static constexpr size_t OW_OFF = V_OFF  + (size_t)64 * 8 * 512 * 64 * 2; // [64][512][512]

// ---------------------------------------------------------------------------
// Kernel 1a: convert weights to f16
// ---------------------------------------------------------------------------
__global__ __launch_bounds__(256) void pack_weights(const float* __restrict__ wqkv,
                                                    const float* __restrict__ wproj,
                                                    _Float16* __restrict__ wq,
                                                    _Float16* __restrict__ wp) {
  int i = blockIdx.x * blockDim.x + threadIdx.x;
  if (i < 512 * 1536) wq[i] = (_Float16)wqkv[i];
  int j = i - 512 * 1536;
  if (j >= 0 && j < 512 * 512) wp[j] = (_Float16)wproj[j];
}

// ---------------------------------------------------------------------------
// Kernel 1b: gather x into window order and convert once to f16.
//   Xw[win*512 + l][c], one float4 -> 4 f16 per thread (4,194,304 threads).
// ---------------------------------------------------------------------------
__global__ __launch_bounds__(256) void pack_x(const float* __restrict__ x,
                                              _Float16* __restrict__ Xw) {
  int idx = blockIdx.x * blockDim.x + threadIdx.x;   // 64 * 512 * 128
  int c4  = idx & 127;
  int l   = (idx >> 7) & 511;
  int win = idx >> 16;
  const float4 v = ((const float4*)(x + (size_t)global_token(win, l) * 512))[c4];
  v4h o;
  o[0] = (_Float16)v.x; o[1] = (_Float16)v.y;
  o[2] = (_Float16)v.z; o[3] = (_Float16)v.w;
  *(v4h*)(Xw + ((size_t)win * 512 + l) * 512 + c4 * 4) = o;
}

// ---------------------------------------------------------------------------
// Kernel 2: QKV GEMM on pre-packed f16 activations.
//   M = 32768 (window order), N = 1536, K = 512.
//   Block 256 thr = 8 waves; wave tile 32(M) x 64(N); grid (128, 24).
//   Writes Q [wh][l][64], Kt [wh][64][l] (pre-transposed), V [wh][l][64].
// ---------------------------------------------------------------------------
__global__ __launch_bounds__(256) void qkv_gemm(const _Float16* __restrict__ Xw,
                                                const _Float16* __restrict__ wq,
                                                _Float16* __restrict__ Q,
                                                _Float16* __restrict__ Kt,
                                                _Float16* __restrict__ V) {
  int lane = threadIdx.x & 31, wave = threadIdx.x >> 5;
  int m0 = blockIdx.x * 256 + wave * 32;
  int n0 = blockIdx.y * 64;

  v8f c[2][4] = {};
  for (int k0 = 0; k0 < 512; k0 += 32) {
    v16h a0 = load_a_f16(Xw, 512, m0,      k0, lane);
    v16h a1 = load_a_f16(Xw, 512, m0 + 16, k0, lane);
#pragma unroll
    for (int t = 0; t < 4; ++t) {
      v16h b = load_b_f16(wq, 1536, k0, n0 + 16 * t, lane);
      c[0][t] = wmma_f16(a0, b, c[0][t]);
      c[1][t] = wmma_f16(a1, b, c[1][t]);
    }
  }

  int half = lane >> 4, col = lane & 15;
  int win = m0 >> 9;  // 32-row strip never crosses a window (512 % 32 == 0)
#pragma unroll
  for (int s = 0; s < 2; ++s) {
#pragma unroll
    for (int t = 0; t < 4; ++t) {
#pragma unroll
      for (int r = 0; r < 8; ++r) {
        float val = c[s][t][r];
        int l = (m0 + s * 16 + r + 8 * half) & 511;
        int n = n0 + t * 16 + col;
        int sect = n >> 9, head = (n >> 6) & 7, d = n & 63;
        size_t hb = (size_t)(win * 8 + head);
        if (sect == 0)      Q [(hb * 512 + l) * 64 + d]  = (_Float16)val;
        else if (sect == 1) Kt[(hb * 64  + d) * 512 + l] = (_Float16)val;
        else                V [(hb * 512 + l) * 64 + d]  = (_Float16)val;
      }
    }
  }
}

// ---------------------------------------------------------------------------
// Kernel 3: flash attention per (window, head).
//   Grid (8 qtiles, 8 heads, 64 windows), 128 thr = 4 independent waves.
//   Each wave owns a 16-row query strip; iterates 512 keys in blocks of 32.
// ---------------------------------------------------------------------------
__global__ __launch_bounds__(128) void attn_kernel(const _Float16* __restrict__ Q,
                                                   const _Float16* __restrict__ Kt,
                                                   const _Float16* __restrict__ V,
                                                   _Float16* __restrict__ Ow) {
  __shared__ __align__(32) _Float16 pbuf[4][16][32];  // per-wave P tile
  int lane = threadIdx.x & 31, wave = threadIdx.x >> 5;
  int qt = blockIdx.x, head = blockIdx.y, win = blockIdx.z;
  int q0 = qt * 64 + wave * 16;
  size_t hb = (size_t)(win * 8 + head);
  const _Float16* Qb = Q  + hb * 512 * 64;
  const _Float16* Kb = Kt + hb * 64 * 512;
  const _Float16* Vb = V  + hb * 512 * 64;

  // Q strip (16 x 64), softmax scale 1/sqrt(64) folded into Q
  v16h qa0 = load_a_f16(Qb, 64, q0, 0, lane);
  v16h qa1 = load_a_f16(Qb, 64, q0, 32, lane);
#pragma unroll
  for (int i = 0; i < 16; ++i) {
    qa0[i] = qa0[i] * (_Float16)0.125f;
    qa1[i] = qa1[i] * (_Float16)0.125f;
  }

  float mrow[8], lrowv[8];
#pragma unroll
  for (int r = 0; r < 8; ++r) { mrow[r] = -3.0e38f; lrowv[r] = 0.0f; }
  v8f o0 = {}, o1 = {}, o2 = {}, o3 = {};

  int half = lane >> 4, col = lane & 15;
  for (int j0 = 0; j0 < 512; j0 += 32) {
    // S tiles: cols [j0, j0+16) and [j0+16, j0+32), K-dim d = 64
    v8f cL = {}, cR = {};
    cL = wmma_f16(qa0, load_b_f16(Kb, 512,  0, j0,      lane), cL);
    cL = wmma_f16(qa1, load_b_f16(Kb, 512, 32, j0,      lane), cL);
    cR = wmma_f16(qa0, load_b_f16(Kb, 512,  0, j0 + 16, lane), cR);
    cR = wmma_f16(qa1, load_b_f16(Kb, 512, 32, j0 + 16, lane), cR);

    // online softmax over the 32 new columns; row = 16 lanes of one half
#pragma unroll
    for (int r = 0; r < 8; ++r) {
      float sL = cL[r], sR = cR[r];
      float mx = fmaxf(sL, sR);
#pragma unroll
      for (int off = 8; off >= 1; off >>= 1) mx = fmaxf(mx, __shfl_xor(mx, off, 16));
      float mnew  = fmaxf(mrow[r], mx);
      float alpha = __expf(mrow[r] - mnew);
      float pL = __expf(sL - mnew);
      float pR = __expf(sR - mnew);
      float s = pL + pR;
#pragma unroll
      for (int off = 8; off >= 1; off >>= 1) s += __shfl_xor(s, off, 16);
      lrowv[r] = lrowv[r] * alpha + s;
      mrow[r]  = mnew;
      o0[r] *= alpha; o1[r] *= alpha; o2[r] *= alpha; o3[r] *= alpha;
      int mr = r + 8 * half;
      pbuf[wave][mr][col]      = (_Float16)pL;   // C-layout -> LDS
      pbuf[wave][mr][col + 16] = (_Float16)pR;
    }

    // reload P as an A-fragment (16x32) and accumulate O += P * V
    v16h pa = load_a_f16(&pbuf[wave][0][0], 32, 0, 0, lane);
    o0 = wmma_f16(pa, load_b_f16(Vb, 64, j0,  0, lane), o0);
    o1 = wmma_f16(pa, load_b_f16(Vb, 64, j0, 16, lane), o1);
    o2 = wmma_f16(pa, load_b_f16(Vb, 64, j0, 32, lane), o2);
    o3 = wmma_f16(pa, load_b_f16(Vb, 64, j0, 48, lane), o3);
  }

  // normalize and store O in [win][l][head*64+d] so the projection reads rows
#pragma unroll
  for (int r = 0; r < 8; ++r) {
    float inv = 1.0f / lrowv[r];
    int ltok = q0 + r + 8 * half;
    size_t rowoff = ((size_t)win * 512 + ltok) * 512 + head * 64;
    Ow[rowoff +  0 + col] = (_Float16)(o0[r] * inv);
    Ow[rowoff + 16 + col] = (_Float16)(o1[r] * inv);
    Ow[rowoff + 32 + col] = (_Float16)(o2[r] * inv);
    Ow[rowoff + 48 + col] = (_Float16)(o3[r] * inv);
  }
}

// ---------------------------------------------------------------------------
// Kernel 4: output projection + bias + window un-partition scatter (f32 out).
//   M = 32768 (window order), N = 512, K = 512.
//   Block 256 thr = 8 waves; wave tile 32(M) x 64(N); grid (128, 8).
// ---------------------------------------------------------------------------
__global__ __launch_bounds__(256) void proj_gemm(const _Float16* __restrict__ Ow,
                                                 const _Float16* __restrict__ wp,
                                                 const float* __restrict__ bias,
                                                 float* __restrict__ out) {
  int lane = threadIdx.x & 31, wave = threadIdx.x >> 5;
  int m0 = blockIdx.x * 256 + wave * 32;
  int n0 = blockIdx.y * 64;

  v8f c[2][4] = {};
  for (int k0 = 0; k0 < 512; k0 += 32) {
    v16h a0 = load_a_f16(Ow, 512, m0,      k0, lane);
    v16h a1 = load_a_f16(Ow, 512, m0 + 16, k0, lane);
#pragma unroll
    for (int t = 0; t < 4; ++t) {
      v16h b = load_b_f16(wp, 512, k0, n0 + 16 * t, lane);
      c[0][t] = wmma_f16(a0, b, c[0][t]);
      c[1][t] = wmma_f16(a1, b, c[1][t]);
    }
  }

  int half = lane >> 4, col = lane & 15;
  int win = m0 >> 9;
  float bv[4];
#pragma unroll
  for (int t = 0; t < 4; ++t) bv[t] = bias[n0 + t * 16 + col];
#pragma unroll
  for (int s = 0; s < 2; ++s) {
#pragma unroll
    for (int r = 0; r < 8; ++r) {
      int l = (m0 + s * 16 + r + 8 * half) & 511;
      size_t base = (size_t)global_token(win, l) * 512 + n0;
#pragma unroll
      for (int t = 0; t < 4; ++t)
        out[base + t * 16 + col] = c[s][t][r] + bv[t];
    }
  }
}

// ---------------------------------------------------------------------------
// Host launch
// ---------------------------------------------------------------------------
extern "C" void kernel_launch(void* const* d_in, const int* in_sizes, int n_in,
                              void* d_out, int out_size, void* d_ws, size_t ws_size,
                              hipStream_t stream) {
  (void)in_sizes; (void)n_in; (void)out_size; (void)ws_size;
  const float* x      = (const float*)d_in[0];
  const float* w_qkv  = (const float*)d_in[1];
  const float* w_proj = (const float*)d_in[2];
  const float* b_proj = (const float*)d_in[3];
  // d_in[4..6] = T,H,W scalars; sizes are fixed by the reference, ignored.

  char* ws = (char*)d_ws;
  _Float16* wq = (_Float16*)(ws + WQ_OFF);
  _Float16* wp = (_Float16*)(ws + WP_OFF);
  _Float16* Xw = (_Float16*)(ws + XW_OFF);
  _Float16* Q  = (_Float16*)(ws + Q_OFF);
  _Float16* Kt = (_Float16*)(ws + KT_OFF);
  _Float16* V  = (_Float16*)(ws + V_OFF);
  _Float16* Ow = (_Float16*)(ws + OW_OFF);
  float* out = (float*)d_out;

  // 1) weights -> f16; x -> window-order f16 (one-shot, removes all inner-loop cvt)
  pack_weights<<<4096, 256, 0, stream>>>(w_qkv, w_proj, wq, wp);
  pack_x<<<16384, 256, 0, stream>>>(x, Xw);
  // 2) QKV GEMM
  qkv_gemm<<<dim3(128, 24), 256, 0, stream>>>(Xw, wq, Q, Kt, V);
  // 3) flash attention per (window, head)
  attn_kernel<<<dim3(8, 8, 64), 128, 0, stream>>>(Q, Kt, V, Ow);
  // 4) projection + bias + un-window scatter
  proj_gemm<<<dim3(128, 8), 256, 0, stream>>>(Ow, wp, b_proj, out);
}
